// BlockWithCache_11690900979836
// MI455X (gfx1250) — matrix operand order repaired and checked by reading, not compiled, not run on hardware
//
#include <hip/hip_runtime.h>
#include <cstdint>

// ---------------------------------------------------------------------------
// Transformer block (Music-Transformer rel-attention) for MI455X / gfx1250.
// All GEMMs + attention core via v_wmma_f32_16x16x32_bf16 (wave32).
// Tile staging via GLOBAL_LOAD_ASYNC_TO_LDS_B128 (ASYNCcnt) when available.
// ---------------------------------------------------------------------------

#define Bq  4
#define Lq  1024
#define Dq  1024
#define Hq  16
#define HSq 64
#define Mq  (Bq * Lq)

typedef __bf16 bf16x16 __attribute__((ext_vector_type(16)));
typedef __bf16 bf16x8  __attribute__((ext_vector_type(8)));
typedef float  floatx8 __attribute__((ext_vector_type(8)));
typedef unsigned short u16x8 __attribute__((ext_vector_type(8)));
typedef int v4i __attribute__((vector_size(16)));   // matches builtin param v4i32

#define AS1 __attribute__((address_space(1)))
#define AS3 __attribute__((address_space(3)))

#if __has_builtin(__builtin_amdgcn_global_load_async_to_lds_b128)
#define HAVE_ASYNC_LDS 1
#else
#define HAVE_ASYNC_LDS 0
#endif

// Issue one 16B async Global->LDS copy (per-lane), tracked by ASYNCcnt.
__device__ __forceinline__ void async_copy16(const void* gsrc, void* ldst) {
#if HAVE_ASYNC_LDS
  __builtin_amdgcn_global_load_async_to_lds_b128(
      (AS1 v4i*)(void*)const_cast<void*>(gsrc),
      (AS3 v4i*)ldst, 0, 0);
#else
  *(v4i*)ldst = *(const v4i*)gsrc;
#endif
}

__device__ __forceinline__ void wait_async0() {
#if __has_builtin(__builtin_amdgcn_s_wait_asynccnt)
  __builtin_amdgcn_s_wait_asynccnt(0);
#else
  asm volatile("s_wait_asynccnt 0" ::: "memory");
#endif
}

__device__ __forceinline__ unsigned short f2bf(float f) {
  union { float f; unsigned u; } v; v.f = f;
  unsigned u = v.u;
  u += 0x7FFFu + ((u >> 16) & 1u);          // round-to-nearest-even
  return (unsigned short)(u >> 16);
}

// Build a 16-element bf16 fragment from two 16B LDS/global chunks.
// Matches ISA layout: vector elems 0..7 = VGPRs 0..3 (pairs), 8..15 = VGPRs 4..7.
__device__ __forceinline__ bf16x16 ld_frag(const unsigned short* lo,
                                           const unsigned short* hi) {
  bf16x8 a = *(const bf16x8*)lo;
  bf16x8 b = *(const bf16x8*)hi;
  bf16x16 r;
#pragma unroll
  for (int i = 0; i < 8; ++i) { r[i] = a[i]; r[i + 8] = b[i]; }
  return r;
}

// ---------------------------------------------------------------------------
// LayerNorm (f32 in) -> bf16 out
// ---------------------------------------------------------------------------
__global__ __launch_bounds__(256)
void ln_to_bf16(const float* __restrict__ x, const float* __restrict__ w,
                const float* __restrict__ b, unsigned short* __restrict__ out,
                int D) {
  const int row = blockIdx.x;
  const float* xr = x + (size_t)row * D;
  float s = 0.f, s2 = 0.f;
  for (int i = threadIdx.x; i < D; i += blockDim.x) {
    float v = xr[i]; s += v; s2 += v * v;
  }
  __shared__ float red[2][8];
#pragma unroll
  for (int off = 16; off; off >>= 1) {
    s  += __shfl_down(s,  off, 32);
    s2 += __shfl_down(s2, off, 32);
  }
  const int wid = threadIdx.x >> 5, lane = threadIdx.x & 31;
  if (lane == 0) { red[0][wid] = s; red[1][wid] = s2; }
  __syncthreads();
  if (threadIdx.x == 0) {
    float a = 0.f, c = 0.f;
#pragma unroll
    for (int i = 0; i < 8; ++i) { a += red[0][i]; c += red[1][i]; }
    red[0][0] = a; red[1][0] = c;
  }
  __syncthreads();
  const float mean = red[0][0] / D;
  const float var  = red[1][0] / D - mean * mean;
  const float inv  = rsqrtf(var + 1e-5f);
  for (int i = threadIdx.x; i < D; i += blockDim.x) {
    float v = (xr[i] - mean) * inv * w[i] + b[i];
    out[(size_t)row * D + i] = f2bf(v);
  }
}

// ---------------------------------------------------------------------------
// W[K][N] f32  ->  Wt[N][K] bf16  (tiled transpose for coalescing)
// ---------------------------------------------------------------------------
__global__ __launch_bounds__(256)
void convert_transpose(const float* __restrict__ W, unsigned short* __restrict__ Wt,
                       int K, int N) {
  __shared__ float tile[32][33];
  const int bn = blockIdx.x * 32, bk = blockIdx.y * 32;
  const int tx = threadIdx.x & 31, ty = threadIdx.x >> 5;   // ty in 0..7
#pragma unroll
  for (int i = 0; i < 32; i += 8)
    tile[ty + i][tx] = W[(size_t)(bk + ty + i) * N + bn + tx];
  __syncthreads();
#pragma unroll
  for (int i = 0; i < 32; i += 8)
    Wt[(size_t)(bn + ty + i) * K + bk + tx] = f2bf(tile[tx][ty + i]);
}

// Er[rows][HS] f32 -> ErP[(rows+pad)][HS] bf16 with zero padding rows
__global__ __launch_bounds__(256)
void convert_er(const float* __restrict__ Er, unsigned short* __restrict__ ErP,
                int rows, int pad, int hs) {
  const int idx = blockIdx.x * blockDim.x + threadIdx.x;
  const int total = (rows + pad) * hs;
  if (idx < total) {
    const int r = idx / hs;
    ErP[idx] = (r < rows) ? f2bf(Er[idx]) : (unsigned short)0;
  }
}

// ---------------------------------------------------------------------------
// GEMM: C[M][N] = A[M][K](bf16) * Bt[N][K](bf16)^T + bias (+gelu) (+res)
// 128x128 block, BK=32, 8 waves (2x4), each wave = 4x2 wmma tiles.
// Tile staging via async Global->LDS B128 copies when the toolchain has them.
// ---------------------------------------------------------------------------
template <int ACT, int OUTBF16>
__global__ __launch_bounds__(256)
void gemm_bf16(const unsigned short* __restrict__ A,
               const unsigned short* __restrict__ Bt,
               const float* __restrict__ bias,
               const float* __restrict__ res,
               void* __restrict__ outv, int M, int N, int K) {
  __shared__ unsigned short As[2][128 * 40];   // 32-deep K, +8 pad
  __shared__ unsigned short Bs[2][128 * 40];
  const int lane = threadIdx.x & 31;
  const int wave = threadIdx.x >> 5;
  const int ln = lane & 15, g = lane >> 4;
  const int waveM = wave >> 2, waveN = wave & 3;
  const size_t gm0 = (size_t)blockIdx.y * 128;
  const size_t gn0 = (size_t)blockIdx.x * 128;
  const int KT = K >> 5;

  auto stage = [&](int buf, int kt) {
#pragma unroll
    for (int c = 0; c < 2; ++c) {
      const int id = threadIdx.x + c * 256;
      const int r = id >> 2, c8 = (id & 3) * 8;
      async_copy16(A  + (gm0 + r) * (size_t)K + (size_t)kt * 32 + c8,
                   &As[buf][r * 40 + c8]);
      async_copy16(Bt + (gn0 + r) * (size_t)K + (size_t)kt * 32 + c8,
                   &Bs[buf][r * 40 + c8]);
    }
  };

  floatx8 acc[4][2] = {};

  stage(0, 0);
  wait_async0();
  __syncthreads();

  for (int kt = 0; kt < KT; ++kt) {
    const int buf = kt & 1;
    if (kt + 1 < KT) stage(buf ^ 1, kt + 1);
    if (kt + 4 < KT) {
      __builtin_prefetch(A  + (gm0 + (threadIdx.x >> 1)) * (size_t)K + (size_t)(kt + 4) * 32, 0, 1);
      __builtin_prefetch(Bt + (gn0 + (threadIdx.x >> 1)) * (size_t)K + (size_t)(kt + 4) * 32, 0, 1);
    }
    bf16x16 af[4], bfr[2];
#pragma unroll
    for (int i = 0; i < 4; ++i) {
      const unsigned short* p = &As[buf][(waveM * 64 + i * 16 + ln) * 40 + g * 16];
      af[i] = ld_frag(p, p + 8);
    }
#pragma unroll
    for (int j = 0; j < 2; ++j) {
      const unsigned short* p = &Bs[buf][(waveN * 32 + j * 16 + ln) * 40 + g * 16];
      bfr[j] = ld_frag(p, p + 8);
    }
#pragma unroll
    for (int i = 0; i < 4; ++i)
#pragma unroll
      for (int j = 0; j < 2; ++j)
        acc[i][j] = __builtin_amdgcn_wmma_f32_16x16x32_bf16(
            false, af[i], false, bfr[j], (short)0, acc[i][j], false, false);
    if (kt + 1 < KT) wait_async0();
    __syncthreads();
  }

#pragma unroll
  for (int i = 0; i < 4; ++i) {
#pragma unroll
    for (int j = 0; j < 2; ++j) {
      const size_t gm = gm0 + waveM * 64 + i * 16 + 8 * g;
      const size_t gn = gn0 + waveN * 32 + j * 16 + ln;
      const float bc = bias ? bias[gn] : 0.f;
#pragma unroll
      for (int r = 0; r < 8; ++r) {
        float v = acc[i][j][r] + bc;
        if (ACT == 1) v = 0.5f * v * (1.0f + erff(v * 0.70710678118f));
        const size_t idx = (gm + r) * (size_t)N + gn;
        if (res) v += res[idx];
        if (OUTBF16) ((unsigned short*)outv)[idx] = f2bf(v);
        else         ((float*)outv)[idx] = v;
      }
    }
  }
}

// ---------------------------------------------------------------------------
// Flash attention with Music-Transformer relative bias (skew).
// grid = (L/128, B*H), 256 threads. Each wave owns 16 Q rows.
// Srel[l,m] = q[l] . ErP[L-1-l+m]  -> per-wave 16x80 band WMMA + shifted gather.
// ---------------------------------------------------------------------------
__global__ __launch_bounds__(256)
void attn_kernel(const unsigned short* __restrict__ qkv,   // [B*L][3D] bf16
                 const unsigned short* __restrict__ erp,   // [(L+128)][HS] bf16
                 unsigned short* __restrict__ y) {         // [B*L][D] bf16
  __shared__ unsigned short Kt[64 * 72];   // K tile  [kv][hs]
  __shared__ unsigned short Vt[64 * 72];   // V tile transposed [hs][kv]
  __shared__ float QP[8 * 1280];           // per-wave: QEr band (f32) / P tile (bf16)

  const int lane = threadIdx.x & 31;
  const int wave = threadIdx.x >> 5;
  const int ln = lane & 15, g = lane >> 4;
  const int bh = blockIdx.y;
  const int b = bh / Hq, h = bh % Hq;
  const int lt = blockIdx.x * 128;
  const int l0 = lt + wave * 16;
  const int LD3 = 3 * Dq;
  const float scale = 0.125f;             // 1/sqrt(64)

  // Q fragments (A-matrix layout) held in registers for all K tiles
  const size_t rowQ = ((size_t)b * Lq + l0 + ln) * LD3 + h * HSq;
  bf16x16 aq[2];
#pragma unroll
  for (int ks = 0; ks < 2; ++ks) {
    const unsigned short* p = qkv + rowQ + ks * 32 + g * 16;
    aq[ks] = ld_frag(p, p + 8);
  }

  float* QErW = &QP[wave * 1280];               // 16 x 80 f32 (wave private)
  unsigned short* PW = (unsigned short*)QErW;   // 16 x 72 bf16 (reuse, later phase)

  floatx8 accO[4] = {};
  float m_i[8], l_i[8];
#pragma unroll
  for (int r = 0; r < 8; ++r) { m_i[r] = -3.0e38f; l_i[r] = 0.f; }

  const int nkt = (lt + 128) >> 6;
  for (int mt = 0; mt < nkt; ++mt) {
    // ---- stage K (async) and V^T tiles ----
    {
      const int m0 = mt * 64;
#pragma unroll
      for (int c = 0; c < 2; ++c) {
        const int id = threadIdx.x + c * 256;
        const int mm = id >> 3, kc = (id & 7) * 8;
        const size_t base = ((size_t)b * Lq + m0 + mm) * LD3 + h * HSq + kc;
        async_copy16(qkv + base + Dq, &Kt[mm * 72 + kc]);
        u16x8 vv = *(const u16x8*)(qkv + base + 2 * Dq);
#pragma unroll
        for (int e = 0; e < 8; ++e) Vt[(kc + e) * 72 + mm] = vv[e];
      }
      wait_async0();
    }
    __syncthreads();

    // ---- S = Q K^T ----
    floatx8 sacc[4] = {};
#pragma unroll
    for (int ks = 0; ks < 2; ++ks) {
#pragma unroll
      for (int nt = 0; nt < 4; ++nt) {
        const unsigned short* p = &Kt[(nt * 16 + ln) * 72 + ks * 32 + g * 16];
        bf16x16 kf = ld_frag(p, p + 8);
        sacc[nt] = __builtin_amdgcn_wmma_f32_16x16x32_bf16(
            false, aq[ks], false, kf, (short)0, sacc[nt], false, false);
      }
    }

    // ---- QEr band: wave-local 16x80 = Q(16x64) @ ErP(band)^T ----
    const int jbase = Lq - 16 - l0 + mt * 64;
#pragma unroll
    for (int c = 0; c < 5; ++c) {
      floatx8 qe = {};
#pragma unroll
      for (int ks = 0; ks < 2; ++ks) {
        int j = jbase + c * 16 + ln;
        j = j < 0 ? 0 : (j > Lq + 127 ? Lq + 127 : j);
        const unsigned short* p = erp + (size_t)j * HSq + ks * 32 + g * 16;
        bf16x16 ef = ld_frag(p, p + 8);
        qe = __builtin_amdgcn_wmma_f32_16x16x32_bf16(
            false, aq[ks], false, ef, (short)0, qe, false, false);
      }
#pragma unroll
      for (int r = 0; r < 8; ++r)
        QErW[(r + 8 * g) * 80 + c * 16 + ln] = qe[r];
    }

    // ---- add shifted Srel, scale, causal mask ----
#pragma unroll
    for (int nt = 0; nt < 4; ++nt) {
#pragma unroll
      for (int r = 0; r < 8; ++r) {
        const int rit = r + 8 * g;
        const int ml = nt * 16 + ln;
        float s = (sacc[nt][r] + QErW[rit * 80 + ml - rit + 15]) * scale;
        if (mt * 64 + ml > l0 + rit) s = -3.0e38f;
        sacc[nt][r] = s;
      }
    }

    // ---- online softmax (rows live in 16-lane halves) ----
#pragma unroll
    for (int r = 0; r < 8; ++r) {
      float tmax = sacc[0][r];
#pragma unroll
      for (int nt = 1; nt < 4; ++nt) tmax = fmaxf(tmax, sacc[nt][r]);
#pragma unroll
      for (int off = 1; off < 16; off <<= 1)
        tmax = fmaxf(tmax, __shfl_xor(tmax, off, 32));
      const float mnew = fmaxf(m_i[r], tmax);
      const float alpha = __expf(m_i[r] - mnew);
      float rs = 0.f;
#pragma unroll
      for (int nt = 0; nt < 4; ++nt) {
        const float p = __expf(sacc[nt][r] - mnew);
        sacc[nt][r] = p;
        rs += p;
      }
#pragma unroll
      for (int off = 1; off < 16; off <<= 1)
        rs += __shfl_xor(rs, off, 32);
      l_i[r] = l_i[r] * alpha + rs;
      m_i[r] = mnew;
#pragma unroll
      for (int ht = 0; ht < 4; ++ht) accO[ht][r] *= alpha;
    }

    // ---- P through LDS (C-layout -> A-layout), O += P @ V ----
#pragma unroll
    for (int nt = 0; nt < 4; ++nt)
#pragma unroll
      for (int r = 0; r < 8; ++r)
        PW[(r + 8 * g) * 72 + nt * 16 + ln] = f2bf(sacc[nt][r]);

#pragma unroll
    for (int ks = 0; ks < 2; ++ks) {
      const unsigned short* pp = &PW[ln * 72 + ks * 32 + g * 16];
      bf16x16 pf = ld_frag(pp, pp + 8);
#pragma unroll
      for (int ht = 0; ht < 4; ++ht) {
        const unsigned short* pv = &Vt[(ht * 16 + ln) * 72 + ks * 32 + g * 16];
        bf16x16 vf = ld_frag(pv, pv + 8);
        accO[ht] = __builtin_amdgcn_wmma_f32_16x16x32_bf16(
            false, pf, false, vf, (short)0, accO[ht], false, false);
      }
    }
    __syncthreads();
  }

  // ---- write y (bf16) ----
#pragma unroll
  for (int ht = 0; ht < 4; ++ht) {
#pragma unroll
    for (int r = 0; r < 8; ++r) {
      const int row = l0 + r + 8 * g;
      const int col = h * HSq + ht * 16 + ln;
      y[((size_t)b * Lq + row) * Dq + col] = f2bf(accO[ht][r] / l_i[r]);
    }
  }
}

// ---------------------------------------------------------------------------
// Host launcher
// ---------------------------------------------------------------------------
extern "C" void kernel_launch(void* const* d_in, const int* in_sizes, int n_in,
                              void* d_out, int out_size, void* d_ws, size_t ws_size,
                              hipStream_t stream) {
  (void)in_sizes; (void)n_in; (void)out_size; (void)ws_size;
  const float* x     = (const float*)d_in[0];
  const float* ln1w  = (const float*)d_in[1];
  const float* ln1b  = (const float*)d_in[2];
  const float* Wqkv  = (const float*)d_in[3];
  const float* bqkv  = (const float*)d_in[4];
  const float* Wproj = (const float*)d_in[5];
  const float* bproj = (const float*)d_in[6];
  const float* Er    = (const float*)d_in[7];
  const float* ln2w  = (const float*)d_in[8];
  const float* ln2b  = (const float*)d_in[9];
  const float* Wfc   = (const float*)d_in[10];
  const float* bfc   = (const float*)d_in[11];
  const float* Wfc2  = (const float*)d_in[12];
  const float* bfc2  = (const float*)d_in[13];

  char* ws = (char*)d_ws;
  size_t off = 0;
  auto alloc = [&](size_t bytes) -> void* {
    void* p = ws + off;
    off += (bytes + 255) & ~(size_t)255;
    return p;
  };
  unsigned short* WqkvT  = (unsigned short*)alloc((size_t)3072 * 1024 * 2);
  unsigned short* WprojT = (unsigned short*)alloc((size_t)1024 * 1024 * 2);
  unsigned short* WfcT   = (unsigned short*)alloc((size_t)4096 * 1024 * 2);
  unsigned short* Wfc2T  = (unsigned short*)alloc((size_t)1024 * 4096 * 2);
  unsigned short* ErP    = (unsigned short*)alloc((size_t)(Lq + 128) * HSq * 2);
  unsigned short* hbf    = (unsigned short*)alloc((size_t)Mq * Dq * 2);
  unsigned short* qkvbf  = (unsigned short*)alloc((size_t)Mq * 3 * Dq * 2);
  unsigned short* ybf    = (unsigned short*)alloc((size_t)Mq * Dq * 2);
  float*          x2     = (float*)alloc((size_t)Mq * Dq * 4);
  unsigned short* h2bf   = (unsigned short*)alloc((size_t)Mq * Dq * 2);
  unsigned short* m1bf   = (unsigned short*)alloc((size_t)Mq * 4 * Dq * 2);

  // weight / Er conversion to bf16 (weights pre-transposed to [N][K])
  convert_transpose<<<dim3(3072 / 32, 1024 / 32), 256, 0, stream>>>(Wqkv, WqkvT, 1024, 3072);
  convert_transpose<<<dim3(1024 / 32, 1024 / 32), 256, 0, stream>>>(Wproj, WprojT, 1024, 1024);
  convert_transpose<<<dim3(4096 / 32, 1024 / 32), 256, 0, stream>>>(Wfc, WfcT, 1024, 4096);
  convert_transpose<<<dim3(1024 / 32, 4096 / 32), 256, 0, stream>>>(Wfc2, Wfc2T, 4096, 1024);
  convert_er<<<((Lq + 128) * HSq + 255) / 256, 256, 0, stream>>>(Er, ErP, Lq, 128, HSq);

  // h = LN1(x) -> bf16
  ln_to_bf16<<<Mq, 256, 0, stream>>>(x, ln1w, ln1b, hbf, Dq);
  // qkv = h @ Wqkv + bqkv (bf16 out)
  gemm_bf16<0, 1><<<dim3(3072 / 128, Mq / 128), 256, 0, stream>>>(
      hbf, WqkvT, bqkv, nullptr, qkvbf, Mq, 3072, 1024);
  // attention -> y (bf16)
  attn_kernel<<<dim3(Lq / 128, Bq * Hq), 256, 0, stream>>>(qkvbf, ErP, ybf);
  // x2 = x + y @ Wproj + bproj (f32)
  gemm_bf16<0, 0><<<dim3(1024 / 128, Mq / 128), 256, 0, stream>>>(
      ybf, WprojT, bproj, x, x2, Mq, 1024, 1024);
  // h2 = LN2(x2)
  ln_to_bf16<<<Mq, 256, 0, stream>>>(x2, ln2w, ln2b, h2bf, Dq);
  // m1 = gelu(h2 @ Wfc + bfc) (bf16)
  gemm_bf16<1, 1><<<dim3(4096 / 128, Mq / 128), 256, 0, stream>>>(
      h2bf, WfcT, bfc, nullptr, m1bf, Mq, 4096, 1024);
  // out = x2 + m1 @ Wfc2 + bfc2 (f32)
  gemm_bf16<0, 0><<<dim3(1024 / 128, Mq / 128), 256, 0, stream>>>(
      m1bf, Wfc2T, bfc2, x2, d_out, Mq, 1024, 4096);
}
